// VAE_62818191671449
// MI455X (gfx1250) — matrix operand order repaired and checked by reading, not compiled, not run on hardware
//
#include <hip/hip_runtime.h>
#include <hip/hip_bf16.h>

// ---------------------------------------------------------------------------
// Graph-VAE forward for MI455X (gfx1250, wave32, WMMA).
// Heavy matmuls: v_wmma_f32_16x16x32_bf16 (bf16 in, f32 accum).
// Memory-bound (~32 GFLOP vs ~1-2 GB traffic @ 23.3 TB/s), so:
//   - all large intermediates stored bf16,
//   - each wave keeps NT (4 or 6) accumulator tiles and reuses one A fragment
//     for NT WMMAs (cuts activation re-reads by NT x),
//   - weight panel staged in LDS once per block (<=64KB of the 320KB WGP LDS),
//     read back via ds_load_b128,
//   - B fragments preloaded into a register array so one s_wait_dscnt covers
//     all NT loads and the NT WMMAs issue back-to-back.
// ---------------------------------------------------------------------------

#define A_AG   8192
#define DEG    32
#define NP     33          // nodes per star subgraph
#define NN     (A_AG*NP)   // 270336 total nodes (multiple of 128)
#define OBS    96
#define HID    128
#define HD     256
#define ZD     64
#define KPOOL  7

typedef __bf16 bf16;
typedef __attribute__((ext_vector_type(16))) __bf16 v16bf;
typedef __attribute__((ext_vector_type(8)))  __bf16 v8bf;
typedef __attribute__((ext_vector_type(8)))  float  v8f;

__device__ __constant__ static const float RS66 = 0.12309149097933274f; // 1/sqrt(66)

__device__ inline float sigf(float x) { return 1.0f / (1.0f + __expf(-x)); }

// ---------------------------------------------------------------------------
// Weight prep: fp32 W[K x N] row-major  ->  bf16 Wt[N x Kpad] (transposed,
// zero-padded in K so the GEMM K-loop is a multiple of 32).
// ---------------------------------------------------------------------------
__global__ void prep_wt(const float* __restrict__ W, bf16* __restrict__ Wt,
                        int K, int N, int Kpad, int total) {
  int t = blockIdx.x * blockDim.x + threadIdx.x;
  if (t >= total) return;
  int n = t / Kpad, k = t - n * Kpad;
  Wt[t] = (k < K) ? (bf16)W[(size_t)k * N + n] : (bf16)0.0f;
}

// ---------------------------------------------------------------------------
// Gather star-subgraph node features:
// node n = a*33 + j ; j==0 -> agent a ; j in 1..32 -> neighbor (a+j) mod A.
// fp32 -> bf16 conversion fused here.
// ---------------------------------------------------------------------------
__global__ void gather_nodes(const float* __restrict__ data, bf16* __restrict__ X,
                             int F, int total) {
  int t = blockIdx.x * blockDim.x + threadIdx.x;
  if (t >= total) return;
  int c = t % F, n = t / F;
  int a = n / NP, j = n - a * NP;
  int src = (j == 0) ? a : (a + j) % A_AG;
  X[t] = (bf16)data[(size_t)src * F + c];
}

// ---------------------------------------------------------------------------
// WMMA GEMM:  out = act( X[M x K](bf16) @ Wt[N x K]^T(bf16) + bias )
//   grid = (M/128, Nout/(NT*16)), block = 256 (8 waves).
//   Wave w owns M-tile w and NT consecutive 16x16 output tiles; per K-step it
//   loads ONE A fragment, preloads NT B fragments from LDS into registers,
//   then issues NT WMMAs back-to-back.
// A fragment (16-bit A 16x32, contiguous K order; B matches): lane l (hi=l>>4) holds row
//   rowBase+(l&15), K = hi*16 + 0..15  -> two 16B global loads.
// B fragment (32x16): lane l holds col (l&15), K = hi*16 + 0..15 -> one
//   contiguous 32B LDS read (2x ds_load_b128) from the staged panel.
// act: 0=none 1=relu 2=sigmoid. outF / outB optional fp32 / bf16 outputs.
// ---------------------------------------------------------------------------
template <int NT>
__global__ __launch_bounds__(256) void wmma_gemm_bf16(
    const bf16* __restrict__ X, const bf16* __restrict__ Wt,
    const float* __restrict__ bias, float* __restrict__ outF,
    bf16* __restrict__ outB, int M, int K, int Nout, int act) {
  extern __shared__ v16bf smem_v[];          // 32B-aligned dynamic LDS
  bf16* bs = (bf16*)smem_v;                  // [NT*16][K] weight panel

  const int colBase0 = blockIdx.y * (NT * 16);

  // Stage the B panel: contiguous [NT*16 x K] bf16 block copy (b128 chunks).
  {
    const uint4* src = (const uint4*)(Wt + (size_t)colBase0 * K);
    uint4* dst = (uint4*)bs;
    const int chunks = (NT * 16 * K) >> 3;   // 8 bf16 per uint4
    for (int i = threadIdx.x; i < chunks; i += 256) dst[i] = src[i];
  }
  __syncthreads();

  const int wave = threadIdx.x >> 5;
  const int lane = threadIdx.x & 31;
  const int hi   = lane >> 4;
  const int l16  = lane & 15;
  const int rowBase = (blockIdx.x * 8 + wave) * 16;

  const bf16* aRow = X + (size_t)(rowBase + l16) * K + hi * 16;

  v8f acc[NT] = {};
  const int kTiles = K >> 5;
  union { v16bf v; v8bf h[2]; } av;
  for (int kt = 0; kt < kTiles; ++kt) {
    const v8bf* ap = (const v8bf*)(aRow + kt * 32);
    av.h[0] = ap[0];                          // K = hi*16 + 0..7
    av.h[1] = ap[1];                          // K = hi*16 + 8..15
    if (kt + 1 < kTiles)                      // -> global_prefetch_b8
      __builtin_prefetch(aRow + kt * 32 + 32, 0, 1);
    const int koff = kt * 32 + hi * 16;
    // Preload all NT B fragments (batched ds_load_b128, single wait) ...
    v16bf bv[NT];
    #pragma unroll
    for (int nt = 0; nt < NT; ++nt)
      bv[nt] = *(const v16bf*)(bs + (size_t)(nt * 16 + l16) * K + koff);
    // ... then issue the NT WMMAs back-to-back.
    #pragma unroll
    for (int nt = 0; nt < NT; ++nt)
      acc[nt] = __builtin_amdgcn_wmma_f32_16x16x32_bf16(
          /*neg_a=*/false, av.v, /*neg_b=*/false, bv[nt],
          /*c_mod=*/(short)0, acc[nt], /*reuse_a=*/false, /*reuse_b=*/false);
  }

  // C/D layout: VGPR r, lanes0-15 -> M=r ; lanes16-31 -> M=8+r ; N=lane&15
  #pragma unroll
  for (int nt = 0; nt < NT; ++nt) {
    const int col = colBase0 + nt * 16 + l16;
    const float badd = bias ? bias[col] : 0.0f;
    #pragma unroll
    for (int r = 0; r < 8; ++r) {
      int row = rowBase + hi * 8 + r;
      float v = acc[nt][r] + badd;
      if (act == 1)      v = fmaxf(v, 0.0f);
      else if (act == 2) v = sigf(v);
      size_t o = (size_t)row * Nout + col;
      if (outF) outF[o] = v;
      if (outB) outB[o] = (bf16)v;
    }
  }
}

// ---------------------------------------------------------------------------
// GCN star-graph mixing (after H = X@W), with bias + relu, bf16 in/out:
//   leaf   = H_c/sqrt(66) + H_leaf/2        + b
//   center = (sum H_leaf)/sqrt(66) + H_c/33 + b
// ---------------------------------------------------------------------------
__global__ void gcn_combine(const bf16* __restrict__ H, bf16* __restrict__ G,
                            const float* __restrict__ bias, int C) {
  int a = blockIdx.x;
  size_t base = (size_t)a * NP * C;
  for (int c = threadIdx.x; c < C; c += blockDim.x) {
    float hc = (float)H[base + c];
    float b  = bias[c];
    float sum = 0.0f;
    for (int j = 1; j < NP; ++j) {
      float hl = (float)H[base + (size_t)j * C + c];
      sum += hl;
      G[base + (size_t)j * C + c] = (bf16)fmaxf(RS66 * hc + 0.5f * hl + b, 0.0f);
    }
    G[base + c] = (bf16)fmaxf(RS66 * sum + hc * (1.0f / 33.0f) + b, 0.0f);
  }
}

// Per-node pooling score pre-mix: sh[n] = dot(G2[n, 0:256], pool_W)
__global__ void pool_dot(const bf16* __restrict__ G, const float* __restrict__ pW,
                         float* __restrict__ sh, int total) {
  int n = blockIdx.x * blockDim.x + threadIdx.x;
  if (n >= total) return;
  const bf16* g = G + (size_t)n * HD;
  float s = 0.0f;
  #pragma unroll 8
  for (int c = 0; c < HD; ++c) s += (float)g[c] * pW[c];
  sh[n] = s;
}

// SAGPool: mix scores through the star GCN, then per-agent top-7 (ties ->
// lowest index, matching lax.top_k), store index + tanh gate.
__global__ void score_topk(const float* __restrict__ sh, const float* __restrict__ pB,
                           int* __restrict__ topIdx, float* __restrict__ topG) {
  int a = blockIdx.x * blockDim.x + threadIdx.x;
  if (a >= A_AG) return;
  float s[NP];
  const float b = pB[0];
  float hc = sh[(size_t)a * NP];
  float sum = 0.0f;
  for (int j = 1; j < NP; ++j) {
    float hl = sh[(size_t)a * NP + j];
    sum += hl;
    s[j] = RS66 * hc + 0.5f * hl + b;
  }
  s[0] = RS66 * sum + hc * (1.0f / 33.0f) + b;
  bool used[NP];
  for (int j = 0; j < NP; ++j) used[j] = false;
  for (int k = 0; k < KPOOL; ++k) {
    int bi = 0; float bv = -3.0e38f;
    for (int j = 0; j < NP; ++j)
      if (!used[j] && s[j] > bv) { bv = s[j]; bi = j; }
    used[bi] = true;
    topIdx[a * KPOOL + k] = bi;
    topG[a * KPOOL + k]   = tanhf(bv);
  }
}

// Gate selected nodes by tanh(score), global-max || global-mean -> feat[A x 512]
__global__ void feat_kernel(const bf16* __restrict__ G, const int* __restrict__ topIdx,
                            const float* __restrict__ topG, bf16* __restrict__ feat) {
  int a = blockIdx.x, c = threadIdx.x;  // blockDim = 256
  float mx = -3.0e38f, mean = 0.0f;
  for (int k = 0; k < KPOOL; ++k) {
    int node = a * NP + topIdx[a * KPOOL + k];
    float v = (float)G[(size_t)node * HD + c] * topG[a * KPOOL + k];
    mx = fmaxf(mx, v);
    mean += v;
  }
  feat[(size_t)a * (2 * HD) + c]      = (bf16)mx;
  feat[(size_t)a * (2 * HD) + HD + c] = (bf16)(mean * (1.0f / KPOOL));
}

// GRU cell gates (gi/gh precomputed by WMMA GEMMs) -> next_hid (fp32 out + bf16)
__global__ void gru_kernel(const float* __restrict__ gi, const float* __restrict__ gh,
                           const float* __restrict__ psi, float* __restrict__ nh,
                           bf16* __restrict__ nhB, int total) {
  int t = blockIdx.x * blockDim.x + threadIdx.x;
  if (t >= total) return;
  int a = t >> 7, c = t & 127;
  size_t o = (size_t)a * (3 * HID) + c;
  float r = sigf(gi[o] + gh[o]);
  float z = sigf(gi[o + HID] + gh[o + HID]);
  float n = tanhf(gi[o + 2 * HID] + r * gh[o + 2 * HID]);
  float p = psi[t];
  float v = (1.0f - z) * n + z * p;
  nh[t]  = v;
  nhB[t] = (bf16)v;
}

// Reparameterize + build padded decoder input [z(64) | pos(2) | zeros(30)] bf16
__global__ void zdin_kernel(const float* __restrict__ mu, const float* __restrict__ lv,
                            const float* __restrict__ eps, const float* __restrict__ pos,
                            bf16* __restrict__ din, int total) {
  int t = blockIdx.x * blockDim.x + threadIdx.x;
  if (t >= total) return;
  int a = t / OBS, c = t - a * OBS;
  float v;
  if (c < ZD)          v = mu[(size_t)a * ZD + c] + __expf(0.5f * lv[(size_t)a * ZD + c]) * eps[(size_t)a * ZD + c];
  else if (c < ZD + 2) v = pos[(size_t)a * 2 + (c - ZD)];
  else                 v = 0.0f;
  din[t] = (bf16)v;
}

// ---------------------------------------------------------------------------
// Host-side orchestration
// ---------------------------------------------------------------------------
static inline int cdiv(int a, int b) { return (a + b - 1) / b; }

extern "C" void kernel_launch(void* const* d_in, const int* in_sizes, int n_in,
                              void* d_out, int out_size, void* d_ws, size_t ws_size,
                              hipStream_t stream) {
  (void)in_sizes; (void)n_in; (void)out_size; (void)ws_size;
  // ---- inputs, in setup_inputs() dict order (params flattened in insertion order)
  const float* obs   = (const float*)d_in[0];
  const float* hid   = (const float*)d_in[1];
  const float* pos   = (const float*)d_in[2];
  const float* eps   = (const float*)d_in[3];
  // obs_net
  const float* o_c1W = (const float*)d_in[4],  *o_c1b = (const float*)d_in[5];
  const float* o_c2W = (const float*)d_in[6],  *o_c2b = (const float*)d_in[7];
  const float* o_pW  = (const float*)d_in[8],  *o_pb  = (const float*)d_in[9];
  const float* o_l1W = (const float*)d_in[10], *o_l1b = (const float*)d_in[11];
  const float* o_l2W = (const float*)d_in[12], *o_l2b = (const float*)d_in[13];
  // hid_net
  const float* h_c1W = (const float*)d_in[14], *h_c1b = (const float*)d_in[15];
  const float* h_c2W = (const float*)d_in[16], *h_c2b = (const float*)d_in[17];
  const float* h_pW  = (const float*)d_in[18], *h_pb  = (const float*)d_in[19];
  const float* h_l1W = (const float*)d_in[20], *h_l1b = (const float*)d_in[21];
  const float* h_l2W = (const float*)d_in[22], *h_l2b = (const float*)d_in[23];
  // GRU + VAE head
  const float* gWih = (const float*)d_in[24], *gbih = (const float*)d_in[25];
  const float* gWhh = (const float*)d_in[26], *gbhh = (const float*)d_in[27];
  const float* encW = (const float*)d_in[28], *encb = (const float*)d_in[29];
  const float* muW  = (const float*)d_in[30], *mub  = (const float*)d_in[31];
  const float* lvW  = (const float*)d_in[32], *lvb  = (const float*)d_in[33];
  const float* d1W  = (const float*)d_in[34], *d1b  = (const float*)d_in[35];
  const float* d2W  = (const float*)d_in[36], *d2b  = (const float*)d_in[37];
  // d_in[38] = matrix (adjacency) -- structure is analytic, unused.

  // ---- outputs (tuple concatenated flat, fp32): rec | next_hid | mu | log_var
  float* out  = (float*)d_out;
  float* recO = out;
  float* nhO  = out + (size_t)A_AG * OBS;
  float* muO  = out + (size_t)A_AG * (OBS + HID);
  float* lvO  = out + (size_t)A_AG * (OBS + HID + ZD);

  // ---- workspace bump allocator
  char* w = (char*)d_ws;
  size_t off = 0;
  auto alloc = [&](size_t bytes) -> void* {
    void* p = w + off;
    off += (bytes + 255) & ~(size_t)255;
    return p;
  };

  // transposed bf16 weights
  bf16* o_c1T = (bf16*)alloc((size_t)64 * OBS * 2);
  bf16* o_c2T = (bf16*)alloc((size_t)HD * 64 * 2);
  bf16* o_l1T = (bf16*)alloc((size_t)HD * 512 * 2);
  bf16* o_l2T = (bf16*)alloc((size_t)OBS * HD * 2);
  bf16* h_c1T = (bf16*)alloc((size_t)64 * HID * 2);
  bf16* h_c2T = (bf16*)alloc((size_t)HD * 64 * 2);
  bf16* h_l1T = (bf16*)alloc((size_t)HD * 512 * 2);
  bf16* h_l2T = (bf16*)alloc((size_t)HID * HD * 2);
  bf16* WihT  = (bf16*)alloc((size_t)384 * OBS * 2);
  bf16* WhhT  = (bf16*)alloc((size_t)384 * HID * 2);
  bf16* encT  = (bf16*)alloc((size_t)HD * HID * 2);
  bf16* muT   = (bf16*)alloc((size_t)ZD * HD * 2);
  bf16* lvT   = (bf16*)alloc((size_t)ZD * HD * 2);
  bf16* d1T   = (bf16*)alloc((size_t)HD * OBS * 2);   // K=66 padded to 96
  bf16* d2T   = (bf16*)alloc((size_t)OBS * HD * 2);

  // activations (reused across the two neigh_nets, processed sequentially)
  bf16* Xn   = (bf16*)alloc((size_t)NN * HID * 2);    // max(F) = 128
  bf16* H64  = (bf16*)alloc((size_t)NN * 64 * 2);
  bf16* G64  = (bf16*)alloc((size_t)NN * 64 * 2);
  bf16* H256 = (bf16*)alloc((size_t)NN * HD * 2);
  bf16* G256 = (bf16*)alloc((size_t)NN * HD * 2);
  float* shS = (float*)alloc((size_t)NN * 4);
  int*  topI = (int*)  alloc((size_t)A_AG * KPOOL * 4);
  float* topG= (float*)alloc((size_t)A_AG * KPOOL * 4);
  bf16* feat = (bf16*)alloc((size_t)A_AG * 512 * 2);
  bf16* h1   = (bf16*)alloc((size_t)A_AG * HD * 2);
  bf16* phiB = (bf16*)alloc((size_t)A_AG * OBS * 2);
  bf16* psiB = (bf16*)alloc((size_t)A_AG * HID * 2);
  float* psiF= (float*)alloc((size_t)A_AG * HID * 4);
  float* giF = (float*)alloc((size_t)A_AG * 384 * 4);
  float* ghF = (float*)alloc((size_t)A_AG * 384 * 4);
  bf16* nhB  = (bf16*)alloc((size_t)A_AG * HID * 2);
  bf16* latB = (bf16*)alloc((size_t)A_AG * HD * 2);
  bf16* dinB = (bf16*)alloc((size_t)A_AG * OBS * 2);
  bf16* decH = (bf16*)alloc((size_t)A_AG * HD * 2);

  auto prep = [&](const float* W, bf16* Wt, int K, int N, int Kpad) {
    int total = N * Kpad;
    prep_wt<<<cdiv(total, 256), 256, 0, stream>>>(W, Wt, K, N, Kpad, total);
  };
  // NT=4 (64-wide panels) when Nout % 64 == 0, NT=6 (96-wide) for Nout=96.
  auto gemm = [&](const bf16* X, const bf16* Wt, const float* bias,
                  float* oF, bf16* oB, int M, int K, int N, int act) {
    if (N % 64 == 0) {
      dim3 grid(M / 128, N / 64);
      size_t smem = (size_t)64 * K * 2;
      wmma_gemm_bf16<4><<<grid, 256, smem, stream>>>(X, Wt, bias, oF, oB, M, K, N, act);
    } else {
      dim3 grid(M / 128, N / 96);
      size_t smem = (size_t)96 * K * 2;
      wmma_gemm_bf16<6><<<grid, 256, smem, stream>>>(X, Wt, bias, oF, oB, M, K, N, act);
    }
  };

  // ---- weight prep (fp32 -> transposed bf16, K padded to multiple of 32)
  prep(o_c1W, o_c1T, OBS, 64, OBS);   prep(o_c2W, o_c2T, 64, HD, 64);
  prep(o_l1W, o_l1T, 512, HD, 512);   prep(o_l2W, o_l2T, HD, OBS, HD);
  prep(h_c1W, h_c1T, HID, 64, HID);   prep(h_c2W, h_c2T, 64, HD, 64);
  prep(h_l1W, h_l1T, 512, HD, 512);   prep(h_l2W, h_l2T, HD, HID, HD);
  prep(gWih, WihT, OBS, 384, OBS);    prep(gWhh, WhhT, HID, 384, HID);
  prep(encW, encT, HID, HD, HID);     prep(muW, muT, HD, ZD, HD);
  prep(lvW, lvT, HD, ZD, HD);         prep(d1W, d1T, ZD + 2, HD, OBS);
  prep(d2W, d2T, HD, OBS, HD);

  // ---- neigh_net (shared pipeline), run for obs then hid
  auto neigh_net = [&](const float* data, int F,
                       const bf16* c1T, const float* c1b,
                       const bf16* c2T, const float* c2b,
                       const float* pW, const float* pb,
                       const bf16* l1T, const float* l1b,
                       const bf16* l2T, const float* l2b,
                       int Fout, float* outF_, bf16* outB_) {
    int tot = NN * F;
    gather_nodes<<<cdiv(tot, 256), 256, 0, stream>>>(data, Xn, F, tot);
    gemm(Xn, c1T, nullptr, nullptr, H64, NN, F, 64, 0);          // c1: X@W
    gcn_combine<<<A_AG, 256, 0, stream>>>(H64, G64, c1b, 64);    // star mix+relu
    gemm(G64, c2T, nullptr, nullptr, H256, NN, 64, HD, 0);       // c2
    gcn_combine<<<A_AG, 256, 0, stream>>>(H256, G256, c2b, HD);
    pool_dot<<<cdiv(NN, 256), 256, 0, stream>>>(G256, pW, shS, NN);
    score_topk<<<cdiv(A_AG, 256), 256, 0, stream>>>(shS, pb, topI, topG);
    feat_kernel<<<A_AG, 256, 0, stream>>>(G256, topI, topG, feat);
    gemm(feat, l1T, l1b, nullptr, h1, A_AG, 512, HD, 1);         // l1 + relu
    gemm(h1, l2T, l2b, outF_, outB_, A_AG, HD, Fout, 0);         // l2
  };

  neigh_net(obs, OBS, o_c1T, o_c1b, o_c2T, o_c2b, o_pW, o_pb,
            o_l1T, o_l1b, o_l2T, o_l2b, OBS, nullptr, phiB);     // phi
  neigh_net(hid, HID, h_c1T, h_c1b, h_c2T, h_c2b, h_pW, h_pb,
            h_l1T, h_l1b, h_l2T, h_l2b, HID, psiF, psiB);        // psi

  // ---- GRU cell
  gemm(phiB, WihT, gbih, giF, nullptr, A_AG, OBS, 384, 0);
  gemm(psiB, WhhT, gbhh, ghF, nullptr, A_AG, HID, 384, 0);
  gru_kernel<<<cdiv(A_AG * HID, 256), 256, 0, stream>>>(giF, ghF, psiF, nhO, nhB,
                                                        A_AG * HID);

  // ---- VAE head
  gemm(nhB, encT, encb, nullptr, latB, A_AG, HID, HD, 0);        // encoder
  gemm(latB, muT, mub, muO, nullptr, A_AG, HD, ZD, 0);           // mu -> d_out
  gemm(latB, lvT, lvb, lvO, nullptr, A_AG, HD, ZD, 0);           // log_var -> d_out
  zdin_kernel<<<cdiv(A_AG * OBS, 256), 256, 0, stream>>>(muO, lvO, eps, pos, dinB,
                                                         A_AG * OBS);
  gemm(dinB, d1T, d1b, nullptr, decH, A_AG, OBS, HD, 1);         // dec1 + relu
  gemm(decH, d2T, d2b, recO, nullptr, A_AG, HD, OBS, 2);         // dec2 + sigmoid
}
